// GATNet_87119116632692
// MI455X (gfx1250) — matrix operand order repaired and checked by reading, not compiled, not run on hardware
//
#include <hip/hip_runtime.h>
#include <hip/hip_bf16.h>
#include <math.h>

// Problem constants (from the reference).
#define Nn   100000
#define Ee   3200000
#define Hh   4
#define Cc   8
#define HC   32          // H*C
#define Gg   512
#define FIN  128
#define NEG_SLOPE 0.2f

typedef __attribute__((ext_vector_type(16))) _Float16 v16h;
typedef __attribute__((ext_vector_type(8)))  _Float16 v8h;
typedef __attribute__((ext_vector_type(8)))  float    v8f;

// ---------------------------------------------------------------- utilities
__global__ __launch_bounds__(256) void k_f32_to_f16(const float* __restrict__ s,
                                                    _Float16* __restrict__ d, long n) {
  long i = (long)blockIdx.x * blockDim.x + threadIdx.x;
  if (i < n) d[i] = (_Float16)s[i];
}

__global__ __launch_bounds__(256) void k_fill(float* __restrict__ p, float v, long n) {
  long i = (long)blockIdx.x * blockDim.x + threadIdx.x;
  if (i < n) p[i] = v;
}

__device__ __forceinline__ void atomicMaxF(float* addr, float v) {
  // Monotone-bits float max: positive floats order as int, negative as reversed uint.
  if (v >= 0.0f) atomicMax((int*)addr, __float_as_int(v));
  else           atomicMin((unsigned int*)addr, (unsigned int)__float_as_int(v));
}

// ---------------------------------------------------------------- WMMA GEMM
// Out[Nn,32] = A[Nn,K](f16, row-major) @ B[K,32](f16, row-major), f32 accum.
// One wave32 computes one 16x16 tile of Out via v_wmma_f32_16x16x32_f16.
//
// A-fragment (ISA 7.12.2, 16-bit A 16x32): lane L<16 holds row M=L,
// K = {k0..k0+7, k0+16..k0+23}; lanes 16-31 same rows, K shifted by +8.
// Two contiguous 16B global loads per lane per K-step.
//
// B-fragment: lane n -> column n&15, K = k0 + 8*(lane>>4) + (j&7) + 16*(j>>3). To avoid
// 16 scalar ds_load_u16 per step, B is staged to LDS with the CDNA5 async
// DMA path (global_load_async_to_lds_b128, ASYNCcnt) and then swizzled once
// into fragment order so the inner loop is two ds_load_b128 per step.
template<int K>
__global__ __launch_bounds__(256) void k_gemm_wmma(const _Float16* __restrict__ A,
                                                   const _Float16* __restrict__ B,
                                                   float* __restrict__ Out) {
  __shared__ __align__(16) _Float16 sRaw[K * HC];   // raw row-major copy of B
  __shared__ __align__(16) _Float16 sBf[K * HC];    // fragment-order B

  // Async DMA: 16B per lane per issue, global -> LDS, no VGPR data movement.
  for (int i = threadIdx.x; i < K * HC / 8; i += 256) {
    unsigned           ldsOff = (unsigned)(size_t)(const void*)(sRaw + i * 8); // addr[31:0] = LDS offset
    unsigned long long gAddr  = (unsigned long long)(const void*)(B + i * 8);
    asm volatile("global_load_async_to_lds_b128 %0, %1, off"
                 :: "v"(ldsOff), "v"(gAddr) : "memory");
  }
  asm volatile("s_wait_asynccnt 0" ::: "memory");
  __syncthreads();

  // One-time swizzle into fragment order:
  // sBf[(((k0/32)*2 + nT)*32 + lane)*16 + j] = B[(k0 + 8*(lane>>4) + (j&7) + 16*(j>>3))*HC + nT*16 + (lane&15)]
  for (int idx = threadIdx.x; idx < K * HC; idx += 256) {
    const int j    = idx & 15;
    const int lane = (idx >> 4) & 31;
    const int t    = idx >> 9;            // (k0/32)*2 + nT
    const int step = t >> 1;
    const int nT   = t & 1;
    const int k    = step * 32 + 8 * (lane >> 4) + (j & 7) + 16 * (j >> 3);
    const int colg = nT * 16 + (lane & 15);
    sBf[idx] = sRaw[k * HC + colg];
  }
  __syncthreads();

  const int lane   = threadIdx.x & 31;
  const int wave   = threadIdx.x >> 5;
  const int tile   = blockIdx.x * 8 + wave;
  const int nTiles = (Nn / 16) * 2;            // 12500
  if (tile >= nTiles) return;
  const int mTile  = tile >> 1;
  const int nTile  = tile & 1;
  const int laneHi = lane >> 4;
  const int lane15 = lane & 15;
  const int row    = mTile * 16 + lane15;      // A row for this lane
  const int col    = nTile * 16 + lane15;      // Out column for this lane

  v8f acc = {};
  #pragma unroll
  for (int k0 = 0; k0 < K; k0 += 32) {
    const _Float16* ap = A + (long)row * K + k0 + laneHi * 8;
    v8h lo = *(const v8h*)(ap);                // K = base .. base+7
    v8h hi = *(const v8h*)(ap + 16);           // K = base+16 .. base+23
    v16h a = __builtin_shufflevector(lo, hi, 0,1,2,3,4,5,6,7,8,9,10,11,12,13,14,15);

    const _Float16* bp = sBf + ((((k0 >> 5) * 2) + nTile) * 32 + lane) * 16;
    v8h b0 = *(const v8h*)(bp);
    v8h b1 = *(const v8h*)(bp + 8);
    v16h b = __builtin_shufflevector(b0, b1, 0,1,2,3,4,5,6,7,8,9,10,11,12,13,14,15);

    acc = __builtin_amdgcn_wmma_f32_16x16x32_f16(false, a, false, b,
                                                 (short)0, acc, false, false);
  }
  // C/D layout: VGPR r -> M = r (lanes 0-15) / r+8 (lanes 16-31), N = lane&15.
  float* op = Out + (long)(mTile * 16 + laneHi * 8) * HC + col;
  #pragma unroll
  for (int r = 0; r < 8; ++r) op[(long)r * HC] = acc[r];
}

// ------------------------------------------------- attention logits (N x H)
__global__ __launch_bounds__(256) void k_attn(const float* __restrict__ h,
                                              const float* __restrict__ a_src,
                                              const float* __restrict__ a_dst,
                                              float* __restrict__ als,
                                              float* __restrict__ ald) {
  long tid = (long)blockIdx.x * blockDim.x + threadIdx.x;
  if (tid >= (long)Nn * Hh) return;
  const int  hd = (int)(tid & 3);
  const long n  = tid >> 2;
  const float* hp = h + n * HC + hd * Cc;
  float s = 0.f, d = 0.f;
  #pragma unroll
  for (int c = 0; c < Cc; ++c) {
    float v = hp[c];
    s += v * a_src[hd * Cc + c];
    d += v * a_dst[hd * Cc + c];
  }
  als[tid] = s;
  ald[tid] = d;
}

// ---------------------------------------------------------------- edge passes
__device__ __forceinline__ void edgeSrcDst(const int* __restrict__ ei, long e,
                                           int& s, int& d) {
  if (e < (long)Ee) { s = ei[e]; d = ei[(long)Ee + e]; }
  else              { s = d = (int)(e - Ee); }            // self-loop
}

__device__ __forceinline__ float edgeLogit(const float* __restrict__ als,
                                           const float* __restrict__ ald,
                                           int s, int d, int hd) {
  float ev = als[s * Hh + hd] + ald[d * Hh + hd];
  return ev > 0.f ? ev : NEG_SLOPE * ev;                  // leaky_relu
}

__global__ __launch_bounds__(256) void k_edge_max(const int* __restrict__ ei,
                                                  const float* __restrict__ als,
                                                  const float* __restrict__ ald,
                                                  float* __restrict__ m) {
  long tid = (long)blockIdx.x * blockDim.x + threadIdx.x;
  if (tid >= (long)(Ee + Nn) * Hh) return;
  const int hd = (int)(tid & 3);
  const long e = tid >> 2;
  if (e + 4096 < (long)Ee) __builtin_prefetch(ei + e + 4096, 0, 0);
  int s, d; edgeSrcDst(ei, e, s, d);
  atomicMaxF(&m[d * Hh + hd], edgeLogit(als, ald, s, d, hd));
}

__global__ __launch_bounds__(256) void k_edge_den(const int* __restrict__ ei,
                                                  const float* __restrict__ als,
                                                  const float* __restrict__ ald,
                                                  const float* __restrict__ m,
                                                  float* __restrict__ denom) {
  long tid = (long)blockIdx.x * blockDim.x + threadIdx.x;
  if (tid >= (long)(Ee + Nn) * Hh) return;
  const int hd = (int)(tid & 3);
  const long e = tid >> 2;
  if (e + 4096 < (long)Ee) __builtin_prefetch(ei + e + 4096, 0, 0);
  int s, d; edgeSrcDst(ei, e, s, d);
  float ev = edgeLogit(als, ald, s, d, hd);
  atomicAdd(&denom[d * Hh + hd], __expf(ev - m[d * Hh + hd]));
}

__global__ __launch_bounds__(256) void k_edge_msg(const int* __restrict__ ei,
                                                  const float* __restrict__ als,
                                                  const float* __restrict__ ald,
                                                  const float* __restrict__ m,
                                                  const float* __restrict__ denom,
                                                  const float* __restrict__ h,
                                                  float* __restrict__ aggr) {
  long tid = (long)blockIdx.x * blockDim.x + threadIdx.x;
  if (tid >= (long)(Ee + Nn) * Hh) return;
  const int hd = (int)(tid & 3);
  const long e = tid >> 2;
  if (e + 4096 < (long)Ee) __builtin_prefetch(ei + e + 4096, 0, 0);
  int s, d; edgeSrcDst(ei, e, s, d);
  float ev    = edgeLogit(als, ald, s, d, hd);
  float alpha = __expf(ev - m[d * Hh + hd]) / denom[d * Hh + hd];
  const float4* hp = (const float4*)(h + (long)s * HC + hd * Cc);
  float4 h0 = hp[0], h1 = hp[1];
  float* ap = aggr + (long)d * HC + hd * Cc;
  atomicAdd(ap + 0, alpha * h0.x); atomicAdd(ap + 1, alpha * h0.y);
  atomicAdd(ap + 2, alpha * h0.z); atomicAdd(ap + 3, alpha * h0.w);
  atomicAdd(ap + 4, alpha * h1.x); atomicAdd(ap + 5, alpha * h1.y);
  atomicAdd(ap + 6, alpha * h1.z); atomicAdd(ap + 7, alpha * h1.w);
}

// ------------------------------------------------------------- bias + ELU
__global__ __launch_bounds__(256) void k_bias_elu(const float* __restrict__ aggr,
                                                  const float* __restrict__ bias,
                                                  float* __restrict__ outF32,
                                                  _Float16* __restrict__ outF16) {
  long tid = (long)blockIdx.x * blockDim.x + threadIdx.x;
  if (tid >= (long)Nn * HC) return;
  const int c = (int)(tid & 31);
  float v = aggr[tid] + bias[c];
  v = v > 0.f ? v : expm1f(v);                 // elu
  outF32[tid] = v;
  outF16[tid] = (_Float16)v;
}

// ------------------------------------------------------------- pooling + FC
__global__ __launch_bounds__(256) void k_pool_accum(const float* __restrict__ h,
                                                    const int* __restrict__ batch,
                                                    float* __restrict__ gsum,
                                                    float* __restrict__ gcnt) {
  long tid = (long)blockIdx.x * blockDim.x + threadIdx.x;
  if (tid >= (long)Nn * HC) return;
  const int  c = (int)(tid & 31);
  const long n = tid >> 5;
  const int  g = batch[n];
  atomicAdd(&gsum[(long)g * HC + c], h[tid]);
  if (c == 0) atomicAdd(&gcnt[g], 1.0f);
}

__global__ __launch_bounds__(256) void k_pool_fc(const float* __restrict__ gsum,
                                                 const float* __restrict__ gcnt,
                                                 const float* __restrict__ fcw,
                                                 const float* __restrict__ fcb,
                                                 float* __restrict__ out) {
  int g = blockIdx.x * blockDim.x + threadIdx.x;
  if (g >= Gg) return;
  float s = 0.f;
  #pragma unroll
  for (int c = 0; c < HC; ++c) s += gsum[(long)g * HC + c] * fcw[c];
  out[g] = s / fmaxf(gcnt[g], 1.0f) + fcb[0];
}

// ---------------------------------------------------------------- launcher
static inline int blocksFor(long n, int bs) { return (int)((n + bs - 1) / bs); }

extern "C" void kernel_launch(void* const* d_in, const int* in_sizes, int n_in,
                              void* d_out, int out_size, void* d_ws, size_t ws_size,
                              hipStream_t stream) {
  (void)in_sizes; (void)n_in; (void)out_size; (void)ws_size;

  const float* x     = (const float*)d_in[0];
  const int*   ei    = (const int*)d_in[1];      // [2,E]: src row then dst row
  const int*   batch = (const int*)d_in[2];
  const float* W[4]  = {(const float*)d_in[3],  (const float*)d_in[7],
                        (const float*)d_in[11], (const float*)d_in[15]};
  const float* As[4] = {(const float*)d_in[4],  (const float*)d_in[8],
                        (const float*)d_in[12], (const float*)d_in[16]};
  const float* Ad[4] = {(const float*)d_in[5],  (const float*)d_in[9],
                        (const float*)d_in[13], (const float*)d_in[17]};
  const float* Bi[4] = {(const float*)d_in[6],  (const float*)d_in[10],
                        (const float*)d_in[14], (const float*)d_in[18]};
  const float* fcw   = (const float*)d_in[19];
  const float* fcb   = (const float*)d_in[20];
  float*       out   = (float*)d_out;

  // Workspace carve-out (~64 MB, all 16B aligned).
  char* ws = (char*)d_ws;
  size_t off = 0;
  _Float16* xf16 = (_Float16*)(ws + off); off += (size_t)Nn * FIN * 2;   // 25.6 MB
  _Float16* wf16 = (_Float16*)(ws + off); off += (size_t)FIN * HC * 2;   // 8 KB
  float*    hf32 = (float*)   (ws + off); off += (size_t)Nn * HC * 4;    // 12.8 MB
  float*    als  = (float*)   (ws + off); off += (size_t)Nn * Hh * 4;
  float*    ald  = (float*)   (ws + off); off += (size_t)Nn * Hh * 4;
  float*    mMax = (float*)   (ws + off); off += (size_t)Nn * Hh * 4;
  float*    dnm  = (float*)   (ws + off); off += (size_t)Nn * Hh * 4;
  float*    aggr = (float*)   (ws + off); off += (size_t)Nn * HC * 4;    // 12.8 MB
  _Float16* hf16 = (_Float16*)(ws + off); off += (size_t)Nn * HC * 2;    // 6.4 MB
  float*    gsum = (float*)   (ws + off); off += (size_t)Gg * HC * 4;
  float*    gcnt = (float*)   (ws + off); off += (size_t)Gg * 4;

  const long nNH   = (long)Nn * Hh;
  const long nNHC  = (long)Nn * HC;
  const long nEH   = (long)(Ee + Nn) * Hh;
  const int  gemmB = ((Nn / 16) * 2 + 7) / 8;                // 1563 blocks

  // One-time input cast to f16 for the WMMA GEMM.
  k_f32_to_f16<<<blocksFor((long)Nn * FIN, 256), 256, 0, stream>>>(x, xf16, (long)Nn * FIN);

  for (int L = 0; L < 4; ++L) {
    const int K = (L == 0) ? FIN : HC;
    k_f32_to_f16<<<blocksFor((long)K * HC, 256), 256, 0, stream>>>(W[L], wf16, (long)K * HC);

    if (K == FIN) k_gemm_wmma<FIN><<<gemmB, 256, 0, stream>>>(xf16, wf16, hf32);
    else          k_gemm_wmma<HC> <<<gemmB, 256, 0, stream>>>(hf16, wf16, hf32);

    k_attn<<<blocksFor(nNH, 256), 256, 0, stream>>>(hf32, As[L], Ad[L], als, ald);

    k_fill<<<blocksFor(nNH, 256), 256, 0, stream>>>(mMax, -INFINITY, nNH);
    k_fill<<<blocksFor(nNH, 256), 256, 0, stream>>>(dnm, 0.0f, nNH);
    k_fill<<<blocksFor(nNHC, 256), 256, 0, stream>>>(aggr, 0.0f, nNHC);

    k_edge_max<<<blocksFor(nEH, 256), 256, 0, stream>>>(ei, als, ald, mMax);
    k_edge_den<<<blocksFor(nEH, 256), 256, 0, stream>>>(ei, als, ald, mMax, dnm);
    k_edge_msg<<<blocksFor(nEH, 256), 256, 0, stream>>>(ei, als, ald, mMax, dnm, hf32, aggr);

    k_bias_elu<<<blocksFor(nNHC, 256), 256, 0, stream>>>(aggr, Bi[L], hf32, hf16);
  }

  // Global mean pool + FC head.
  k_fill<<<blocksFor((long)Gg * HC, 256), 256, 0, stream>>>(gsum, 0.0f, (long)Gg * HC);
  k_fill<<<blocksFor((long)Gg, 256), 256, 0, stream>>>(gcnt, 0.0f, (long)Gg);
  k_pool_accum<<<blocksFor(nNHC, 256), 256, 0, stream>>>(hf32, batch, gsum, gcnt);
  k_pool_fc<<<blocksFor((long)Gg, 256), 256, 0, stream>>>(gsum, gcnt, fcw, fcb, out);
}
